// CrossModalAttention_63007170232767
// MI455X (gfx1250) — matrix operand order repaired and checked by reading, not compiled, not run on hardware
//
#include <hip/hip_runtime.h>
#include <hip/hip_bf16.h>

typedef __attribute__((ext_vector_type(16))) __bf16 v16bf;
typedef __attribute__((ext_vector_type(8)))  __bf16 v8bf;
typedef __attribute__((ext_vector_type(8)))  float  v8f;

#define B_   2
#define NQ   2048
#define NKV  2048
#define D_   512
#define H_   8
#define HD_  64
#define MPROJ (B_*NQ)   // 4096 rows per projection

// ---- fragment helpers (CDNA5 wave32 WMMA layouts, ISA 7.12.2) -------------
static __device__ __forceinline__ v8f zerov() { v8f z = {}; return z; }

static __device__ __forceinline__ v16bf cat16(v8bf lo, v8bf hi) {
  return __builtin_shufflevector(lo, hi, 0,1,2,3,4,5,6,7,8,9,10,11,12,13,14,15);
}
// B fragment: lane holds 16 consecutive K for fixed N
static __device__ __forceinline__ v16bf load16(const __bf16* p) {
  return cat16(*(const v8bf*)p, *(const v8bf*)(p + 8));
}
// A fragment: lane l%16 = row M; half hi: K = ks+16*hi+0..15
static __device__ __forceinline__ v16bf loadA(const __bf16* rowp, int ks, int hi) {
  return cat16(*(const v8bf*)(rowp + ks + 16*hi),
               *(const v8bf*)(rowp + ks + 16*hi + 8));
}
static __device__ __forceinline__ v8f wmma_bf16(v16bf a, v16bf b, v8f c) {
  return __builtin_amdgcn_wmma_f32_16x16x32_bf16(false, a, false, b,
                                                 (short)0, c, false, false);
}

// ---- CDNA5 async global->LDS copy (ASYNCcnt path, ISA ch.8 / 15.18.3) -----
// Generic pointers to LDS carry the wave-relative LDS byte offset in the low
// 32 bits (flat aperture rule).
static __device__ __forceinline__ void async_copy_b128(void* lds_dst, const void* gsrc) {
  unsigned int off = (unsigned int)(uintptr_t)lds_dst;
  asm volatile("global_load_async_to_lds_b128 %0, %1, off"
               :: "v"(off), "v"(gsrc) : "memory");
}
static __device__ __forceinline__ void async_wait0() {
  asm volatile("s_wait_asynccnt 0x0" ::: "memory");
}

// ---- fp32 -> bf16 conversion kernels --------------------------------------
__global__ void cvt_act(const float* __restrict__ in, __bf16* __restrict__ outp, int n) {
  int i = blockIdx.x * blockDim.x + threadIdx.x;
  int stride = gridDim.x * blockDim.x;
  for (; i < n; i += stride) outp[i] = (__bf16)in[i];
}

// W[k][n] fp32 -> Wt[n][k] bf16 (so GEMM B-fragments are K-contiguous)
__global__ void cvt_wtr(const float* __restrict__ W, __bf16* __restrict__ Wt) {
  int i = blockIdx.x * blockDim.x + threadIdx.x;     // i = n*512 + k
  if (i < D_ * D_) {
    int n = i >> 9, k = i & 511;
    Wt[i] = (__bf16)W[(size_t)k * D_ + n];
  }
}

// ---- projection GEMM: C[M=4096][N=512] = X*W + bias, bf16 WMMA ------------
// transposeOut=0 -> out[b][h][s][hd]; transposeOut=1 -> out[b][h][hd][s]
__global__ __launch_bounds__(256) void proj_gemm(
    const __bf16* __restrict__ X,   // [M][512] row major
    const __bf16* __restrict__ Wt,  // [n][k] row major
    const float*  __restrict__ bias,
    __bf16* __restrict__ out, int transposeOut)
{
  __shared__ __align__(16) __bf16 As[128][32];
  __shared__ __align__(16) __bf16 Bs[128][32];

  const int t = threadIdx.x;
  const int wv = t >> 5, lane = t & 31, lm = lane & 15, hi = lane >> 4;
  const int waveM = (wv >> 2) * 64, waveN = (wv & 3) * 32;   // 2x4 wave grid
  const int m0 = blockIdx.x * 128, n0 = blockIdx.y * 128;

  v8f acc[4][2];
  #pragma unroll
  for (int mi = 0; mi < 4; ++mi) { acc[mi][0] = zerov(); acc[mi][1] = zerov(); }

  for (int kt = 0; kt < D_; kt += 32) {
    __syncthreads();
    {   // async DMA stage of A and B tiles straight into LDS (no VGPR bounce)
      const int row = t >> 1, kh = (t & 1) * 16;
      #pragma unroll
      for (int j = 0; j < 2; ++j) {
        async_copy_b128(&As[row][kh + j*8],
                        &X [(size_t)(m0 + row) * D_ + kt + kh + j*8]);
        async_copy_b128(&Bs[row][kh + j*8],
                        &Wt[(size_t)(n0 + row) * D_ + kt + kh + j*8]);
      }
      async_wait0();
    }
    __syncthreads();

    v16bf a[4], bb[2];
    #pragma unroll
    for (int mi = 0; mi < 4; ++mi) a[mi] = loadA(&As[waveM + mi*16 + lm][0], 0, hi);
    #pragma unroll
    for (int ni = 0; ni < 2; ++ni) bb[ni] = load16(&Bs[waveN + ni*16 + lm][16*hi]);
    #pragma unroll
    for (int mi = 0; mi < 4; ++mi)
      #pragma unroll
      for (int ni = 0; ni < 2; ++ni)
        acc[mi][ni] = wmma_bf16(a[mi], bb[ni], acc[mi][ni]);
  }

  // epilogue: +bias, cast, scatter into head-major (optionally transposed) layout
  #pragma unroll
  for (int ni = 0; ni < 2; ++ni) {
    const int gn = n0 + waveN + ni*16 + lm;
    const float bv = bias[gn];
    const int hh = (gn >> 6) & 7, hd = gn & 63;
    #pragma unroll
    for (int mi = 0; mi < 4; ++mi) {
      #pragma unroll
      for (int r = 0; r < 8; ++r) {
        const int gm = m0 + waveM + mi*16 + r + 8*hi;     // row = r + 8*hi of tile
        const int bidx = gm >> 11, srow = gm & 2047;
        size_t idx = transposeOut
            ? ((((size_t)bidx*H_ + hh)*HD_ + hd)*NKV + srow)
            : ((((size_t)bidx*H_ + hh)*NQ  + srow)*HD_ + hd);
        out[idx] = (__bf16)(acc[mi][ni][r] + bv);
      }
    }
  }
}

// ---- flash attention over the unmasked (key_value) keys -------------------
// grid = B*H*(NQ/64), block = 128 (4 waves, 16 Q rows each)
__global__ __launch_bounds__(128) void attn_kernel(
    const __bf16* __restrict__ Qh,  // [b][h][s][hd]
    const __bf16* __restrict__ Kh,  // [b][h][s][hd]
    const __bf16* __restrict__ Vt,  // [b][h][hd][s]
    float* __restrict__ ctx)        // [b][s][d]
{
  __shared__ __align__(16) __bf16 Ks[64][64];      // [key][d]
  __shared__ __align__(16) __bf16 Vs[64][64];      // [d][key]
  __shared__ __align__(16) __bf16 Ps[4][16][64];   // per-wave P re-layout

  const int t = threadIdx.x;
  const int wave = t >> 5, lane = t & 31, lm = lane & 15, hi = lane >> 4;

  int bid = blockIdx.x;
  const int qt = bid & 31; bid >>= 5;
  const int h  = bid & 7;  bid >>= 3;
  const int b  = bid;

  // persistent Q A-fragments (16 rows x 64 d, two 32-wide K windows)
  const __bf16* Qrow =
      Qh + (((size_t)(b*H_ + h))*NQ + (size_t)qt*64 + wave*16 + lm) * HD_;
  const v16bf aQ0 = loadA(Qrow, 0, hi);
  const v16bf aQ1 = loadA(Qrow, 32, hi);

  v8f O0 = {}, O1 = {}, O2 = {}, O3 = {};
  float runm[8], runl[8];
  #pragma unroll
  for (int r = 0; r < 8; ++r) { runm[r] = -1e30f; runl[r] = 0.f; }

  const __bf16* Kbase = Kh + ((size_t)(b*H_ + h)) * NKV * HD_;
  const __bf16* Vbase = Vt + ((size_t)(b*H_ + h)) * HD_ * NKV;

  for (int kv0 = 0; kv0 < NKV; kv0 += 64) {
    __syncthreads();
    {   // async DMA stage of 64x64 K and V tiles (64B per thread each)
      const int r = t >> 1, c0 = (t & 1) * 32;
      #pragma unroll
      for (int j = 0; j < 4; ++j) {
        async_copy_b128(&Ks[r][c0 + j*8],
                        &Kbase[(size_t)(kv0 + r)*HD_ + c0 + j*8]);
        async_copy_b128(&Vs[r][c0 + j*8],
                        &Vbase[(size_t)r*NKV + kv0 + c0 + j*8]);
      }
      // prefetch next KV tile into L2 while we compute on this one
      if (kv0 + 64 < NKV) {
        __builtin_prefetch(&Kbase[(size_t)(kv0 + 64 + r)*HD_ + c0], 0, 3);
        __builtin_prefetch(&Vbase[(size_t)r*NKV + kv0 + 64 + c0], 0, 3);
      }
      async_wait0();
    }
    __syncthreads();

    // S = Q * K^T   (4 N-tiles of 16 keys, Kdim=64 in two WMMA steps)
    v8f sf[4];
    #pragma unroll
    for (int nk = 0; nk < 4; ++nk) {
      v8f a = zerov();
      a = wmma_bf16(aQ0, load16(&Ks[nk*16 + lm][16*hi]),      a);
      a = wmma_bf16(aQ1, load16(&Ks[nk*16 + lm][32 + 16*hi]), a);
      sf[nk] = a;
    }

    // online softmax: row r+8*hi lives in slot r across the 16-lane half
    #pragma unroll
    for (int r = 0; r < 8; ++r) {
      sf[0][r] *= 0.125f; sf[1][r] *= 0.125f;
      sf[2][r] *= 0.125f; sf[3][r] *= 0.125f;
      float m = fmaxf(fmaxf(sf[0][r], sf[1][r]), fmaxf(sf[2][r], sf[3][r]));
      #pragma unroll
      for (int msk = 1; msk < 16; msk <<= 1)
        m = fmaxf(m, __shfl_xor(m, msk, 32));
      const float nm = fmaxf(runm[r], m);
      const float corr = __expf(runm[r] - nm);
      runm[r] = nm;
      runl[r] *= corr;
      O0[r] *= corr; O1[r] *= corr; O2[r] *= corr; O3[r] *= corr;
    }

    // P = exp(S - m), re-layout through per-wave LDS for A-fragments
    #pragma unroll
    for (int nk = 0; nk < 4; ++nk) {
      #pragma unroll
      for (int r = 0; r < 8; ++r) {
        const float p = __expf(sf[nk][r] - runm[r]);
        runl[r] += p;
        Ps[wave][r + 8*hi][nk*16 + lm] = (__bf16)p;
      }
    }

    // O += P * V
    #pragma unroll
    for (int ksi = 0; ksi < 2; ++ksi) {
      const v16bf aP = loadA(&Ps[wave][lm][0], ksi*32, hi);
      O0 = wmma_bf16(aP, load16(&Vs[ 0 + lm][ksi*32 + 16*hi]), O0);
      O1 = wmma_bf16(aP, load16(&Vs[16 + lm][ksi*32 + 16*hi]), O1);
      O2 = wmma_bf16(aP, load16(&Vs[32 + lm][ksi*32 + 16*hi]), O2);
      O3 = wmma_bf16(aP, load16(&Vs[48 + lm][ksi*32 + 16*hi]), O3);
    }
  }

  // finish row sums across the 16-lane half, normalize, write ctx (fp32)
  #pragma unroll
  for (int r = 0; r < 8; ++r) {
    #pragma unroll
    for (int msk = 1; msk < 16; msk <<= 1)
      runl[r] += __shfl_xor(runl[r], msk, 32);
  }
  #pragma unroll
  for (int r = 0; r < 8; ++r) {
    const int srow = qt*64 + wave*16 + r + 8*hi;
    const float inv = 1.0f / runl[r];
    float* crow = ctx + ((size_t)b*NQ + srow)*D_ + h*HD_;
    crow[ 0 + lm] = O0[r] * inv;
    crow[16 + lm] = O1[r] * inv;
    crow[32 + lm] = O2[r] * inv;
    crow[48 + lm] = O3[r] * inv;
  }
}

// ---- residual + LayerNorm --------------------------------------------------
__global__ __launch_bounds__(256) void resid_ln(
    const float* __restrict__ q, const float* __restrict__ ctx,
    const float* __restrict__ gamma, const float* __restrict__ beta,
    float* __restrict__ out)
{
  __shared__ float ssum[256], ssq[256];
  const int row = blockIdx.x, t = threadIdx.x;
  const size_t base = (size_t)row * D_;
  const float x0 = q[base + t]       + ctx[base + t];
  const float x1 = q[base + 256 + t] + ctx[base + 256 + t];
  ssum[t] = x0 + x1;
  ssq[t]  = x0*x0 + x1*x1;
  __syncthreads();
  for (int off = 128; off > 0; off >>= 1) {
    if (t < off) { ssum[t] += ssum[t + off]; ssq[t] += ssq[t + off]; }
    __syncthreads();
  }
  const float mu  = ssum[0] * (1.0f / 512.0f);
  const float var = ssq[0]  * (1.0f / 512.0f) - mu * mu;
  const float rs  = rsqrtf(var + 1e-5f);
  out[base + t]       = (x0 - mu) * rs * gamma[t]       + beta[t];
  out[base + 256 + t] = (x1 - mu) * rs * gamma[256 + t] + beta[256 + t];
}

// ---- host ------------------------------------------------------------------
extern "C" void kernel_launch(void* const* d_in, const int* in_sizes, int n_in,
                              void* d_out, int out_size, void* d_ws, size_t ws_size,
                              hipStream_t stream) {
  const float* query = (const float*)d_in[0];
  const float* keyv  = (const float*)d_in[1];
  const float* Wq    = (const float*)d_in[2];
  const float* bq    = (const float*)d_in[3];
  const float* Wk    = (const float*)d_in[4];
  const float* bk    = (const float*)d_in[5];
  const float* Wv    = (const float*)d_in[6];
  const float* bv    = (const float*)d_in[7];
  const float* gamma = (const float*)d_in[8];
  const float* beta  = (const float*)d_in[9];
  float* outp = (float*)d_out;

  char* ws = (char*)d_ws;
  auto take = [&](size_t bytes) {
    char* p = ws; ws += (bytes + 255) & ~(size_t)255; return p;
  };
  __bf16* Xq  = (__bf16*)take((size_t)MPROJ * D_ * 2);
  __bf16* Xkv = (__bf16*)take((size_t)MPROJ * D_ * 2);
  __bf16* Wqt = (__bf16*)take((size_t)D_ * D_ * 2);
  __bf16* Wkt = (__bf16*)take((size_t)D_ * D_ * 2);
  __bf16* Wvt = (__bf16*)take((size_t)D_ * D_ * 2);
  __bf16* Qh  = (__bf16*)take((size_t)MPROJ * D_ * 2);
  __bf16* Kh  = (__bf16*)take((size_t)MPROJ * D_ * 2);
  __bf16* Vt  = (__bf16*)take((size_t)MPROJ * D_ * 2);
  float*  ctx = (float*) take((size_t)MPROJ * D_ * 4);

  cvt_act<<<2048, 256, 0, stream>>>(query, Xq,  MPROJ * D_);
  cvt_act<<<2048, 256, 0, stream>>>(keyv,  Xkv, MPROJ * D_);
  cvt_wtr<<<(D_*D_)/256, 256, 0, stream>>>(Wq, Wqt);
  cvt_wtr<<<(D_*D_)/256, 256, 0, stream>>>(Wk, Wkt);
  cvt_wtr<<<(D_*D_)/256, 256, 0, stream>>>(Wv, Wvt);

  dim3 pg(MPROJ / 128, D_ / 128);
  proj_gemm<<<pg, 256, 0, stream>>>(Xq,  Wqt, bq, Qh, 0);
  proj_gemm<<<pg, 256, 0, stream>>>(Xkv, Wkt, bk, Kh, 0);
  proj_gemm<<<pg, 256, 0, stream>>>(Xkv, Wvt, bv, Vt, 1);

  attn_kernel<<<B_ * H_ * (NQ / 64), 128, 0, stream>>>(Qh, Kh, Vt, ctx);
  resid_ln<<<B_ * NQ, 256, 0, stream>>>(query, ctx, gamma, beta, outp);
}
